// Phi3AttentionExperimental_79439715107432
// MI455X (gfx1250) — compile-verified, matrix operand order checked by reading, not run on hardware
//
#include <hip/hip_runtime.h>

#define H_   32
#define KVH_ 8
#define D_   96
#define HID_ 3072
#define B_   2
#define S_   2048
#define OP_  (H_*D_ + 2*KVH_*D_)   // 4608

typedef __attribute__((ext_vector_type(16))) __bf16          v16bf;
typedef __attribute__((ext_vector_type(8)))  float           v8f;
typedef __attribute__((ext_vector_type(8)))  unsigned short  u16x8;
typedef __attribute__((ext_vector_type(4)))  unsigned short  u16x4;
typedef __attribute__((ext_vector_type(4)))  unsigned int    u32x4;
typedef __attribute__((ext_vector_type(8)))  int             i32x8;
typedef __attribute__((ext_vector_type(4)))  int             i32x4;

struct Frag { u16x8 lo, hi; };

__device__ __forceinline__ unsigned short f2bf(float f) {
  unsigned u = __builtin_bit_cast(unsigned, f);
  u += 0x7fffu + ((u >> 16) & 1u);          // round-to-nearest-even
  return (unsigned short)(u >> 16);
}

// A/B fragment for v_wmma_f32_16x16x32_bf16.
// rowp = start of this lane's row (row-major, contraction dim contiguous).
// elems 0..7 = K(kb..kb+7), elems 8..15 = K(kb+16..kb+23), kb = 8*(lane>=16).
__device__ __forceinline__ v16bf load_frag(const unsigned short* rowp, int kb) {
  Frag t;
  t.lo = *(const u16x8*)(rowp + kb);
  t.hi = *(const u16x8*)(rowp + kb + 16);
  return __builtin_bit_cast(v16bf, t);
}

__device__ __forceinline__ v8f wmma_bf16(v16bf a, v16bf b, v8f c) {
  return __builtin_amdgcn_wmma_f32_16x16x32_bf16(false, a, false, b, (short)0, c, false, false);
}

#if __has_builtin(__builtin_amdgcn_tensor_load_to_lds)
#define HAVE_TDM 1
#else
#define HAVE_TDM 0
#endif

#if HAVE_TDM
// 2D TDM tile load: global (row-major, rowStride elems of bf16) -> LDS.
// Pads 16B per 128B row => LDS row stride 144B (16B aligned, bank-conflict free).
// Descriptor per CDNA5 ISA §8.3-8.5 (D# group0 + group1, groups 2/3 zero).
__device__ __forceinline__ void tdm_load_2d(unsigned lds_byte_off,
                                            const unsigned short* tile,
                                            unsigned rowStride,  // elems
                                            unsigned tileW,      // elems (mult of 64B worth)
                                            unsigned tileH) {
  unsigned long long ga = (unsigned long long)(uintptr_t)tile;
  u32x4 g0;
  g0[0] = 1u;                                      // count=1, user mode
  g0[1] = lds_byte_off;                            // lds_addr
  g0[2] = (unsigned)ga;                            // global_addr[31:0]
  g0[3] = (unsigned)(ga >> 32) | 0x80000000u;      // global_addr[56:32] | type=2
  i32x8 g1;
  g1[0] = (int)((1u << 16)        // data_size = 1 (2 bytes)
              | (1u << 20)        // pad_enable
              | (4u << 22)        // pad_interval: 128B of data
              | (3u << 25));      // pad_amount: 4 DWORDs = 16B
  g1[1] = (int)((tileW & 0xffffu) << 16);                      // tensor_dim0 lo16
  g1[2] = (int)((tileW >> 16) | ((tileH & 0xffffu) << 16));    // dim0 hi / dim1 lo
  g1[3] = (int)((tileH >> 16) | (tileW << 16));                // dim1 hi / tile_dim0
  g1[4] = (int)tileH;                                          // tile_dim1 (tile_dim2=0)
  g1[5] = (int)rowStride;                                      // tensor_dim0_stride lo32
  g1[6] = 0;
  g1[7] = 0;
  i32x4 z4 = {0, 0, 0, 0};
#if defined(__clang_major__) && __clang_major__ >= 23
  i32x8 z8 = {0, 0, 0, 0, 0, 0, 0, 0};
  __builtin_amdgcn_tensor_load_to_lds(g0, g1, z4, z4, z8, 0);
#else
  __builtin_amdgcn_tensor_load_to_lds(g0, g1, z4, z4, 0);
#endif
}
#endif

// ---------------------------------------------------------------------------
// C(MxN, f32) = A(MxK, bf16) @ W(NxK, bf16)^T.
// Block tile 128x256, K-step 64, 256 threads = 8 waves (2x4 grid of 64x64).
// Double-buffered LDS tiles delivered by the Tensor Data Mover.
// LDS row stride = 72 halves (144B: 128B data + 16B TDM pad).
// ---------------------------------------------------------------------------
#define BM 128
#define BN 256
#define BK 64
#define LDST 72
#define A_TILE_HALF (BM * LDST)            // 9216 halves (18432 B)
#define W_TILE_HALF (BN * LDST)            // 18432 halves (36864 B)
#define STAGE_HALF  (A_TILE_HALF + W_TILE_HALF)
#define STAGE_BYTES (STAGE_HALF * 2)       // 55296 B per stage, 2 stages = 108 KB

__global__ __launch_bounds__(256) void gemm_bf16_tdm(
    const unsigned short* __restrict__ A, const unsigned short* __restrict__ W,
    float* __restrict__ C, int M, int N, int K)
{
  extern __shared__ unsigned short smem[];
  (void)M;
  const int t    = threadIdx.x;
  const int lane = t & 31;
  const int wave = t >> 5;
  const int wm   = wave >> 2;        // 0..1  (64-row slab)
  const int wn   = wave & 3;         // 0..3  (64-col slab)
  const int m0   = blockIdx.y * BM;
  const int n0   = blockIdx.x * BN;
  const int arow = lane & 15;
  const int kb   = (lane >> 4) * 8;

  v8f acc[4][4];
#pragma unroll
  for (int mi = 0; mi < 4; ++mi)
#pragma unroll
    for (int ni = 0; ni < 4; ++ni) acc[mi][ni] = {};

#if HAVE_TDM
  const unsigned smem_base = (unsigned)(uintptr_t)smem;
  const bool issuer = (wave == 0);
  if (issuer) {
    tdm_load_2d(smem_base,                    A + (size_t)m0 * K, K, BK, BM);
    tdm_load_2d(smem_base + A_TILE_HALF * 2,  W + (size_t)n0 * K, K, BK, BN);
  }
#endif
  int cur = 0;

  for (int k0 = 0; k0 < K; k0 += BK) {
#if HAVE_TDM
    const bool more = (k0 + BK) < K;
    if (issuer) {
      if (more) {
        const unsigned nb = smem_base + (cur ^ 1) * STAGE_BYTES;
        tdm_load_2d(nb,                    A + (size_t)m0 * K + (k0 + BK), K, BK, BM);
        tdm_load_2d(nb + A_TILE_HALF * 2,  W + (size_t)n0 * K + (k0 + BK), K, BK, BN);
        __builtin_amdgcn_s_wait_tensorcnt(2);   // current stage's 2 loads done
      } else {
        __builtin_amdgcn_s_wait_tensorcnt(0);
      }
    }
    __syncthreads();
#else
    // Cooperative fallback: 256 threads copy tiles into the padded layout.
    for (int i = t; i < BM * 8; i += 256) {
      const int row = i >> 3, ch = (i & 7) * 8;
      *(u16x8*)&smem[row * LDST + ch] =
          *(const u16x8*)(A + (size_t)(m0 + row) * K + k0 + ch);
    }
    for (int i = t; i < BN * 8; i += 256) {
      const int row = i >> 3, ch = (i & 7) * 8;
      *(u16x8*)&smem[A_TILE_HALF + row * LDST + ch] =
          *(const u16x8*)(W + (size_t)(n0 + row) * K + k0 + ch);
    }
    __syncthreads();
#endif
    const unsigned short* As = smem + cur * STAGE_HALF;
    const unsigned short* Ws = As + A_TILE_HALF;

#pragma unroll
    for (int c = 0; c < 2; ++c) {
      v16bf af[4], bf[4];
#pragma unroll
      for (int mi = 0; mi < 4; ++mi)
        af[mi] = load_frag(&As[(wm * 64 + mi * 16 + arow) * LDST + c * 32], kb);
#pragma unroll
      for (int ni = 0; ni < 4; ++ni)
        bf[ni] = load_frag(&Ws[(wn * 64 + ni * 16 + arow) * LDST + c * 32], kb);
#pragma unroll
      for (int mi = 0; mi < 4; ++mi)
#pragma unroll
        for (int ni = 0; ni < 4; ++ni)
          acc[mi][ni] = wmma_bf16(af[mi], bf[ni], acc[mi][ni]);
    }
    __syncthreads();
#if HAVE_TDM
    cur ^= 1;
#endif
  }

  // Epilogue: 32-bit linear indices (max 4096*4608 < 2^31) -> cheap addressing.
  const int rbase = (lane >> 4) * 8;
  const int col   = lane & 15;
#pragma unroll
  for (int mi = 0; mi < 4; ++mi) {
    const int row = m0 + wm * 64 + mi * 16 + rbase;
#pragma unroll
    for (int ni = 0; ni < 4; ++ni) {
      const int idx0 = row * N + (n0 + wn * 64 + ni * 16 + col);
#pragma unroll
      for (int r = 0; r < 8; ++r)
        C[idx0 + r * N] = acc[mi][ni][r];
    }
  }
}

// ---------------------------------------------------------------------------
// fp32 -> bf16 bulk convert (vectorized x4). n4 = elems/4.
// ---------------------------------------------------------------------------
__global__ __launch_bounds__(256) void cvt_f32_bf16(
    const float* __restrict__ s, unsigned short* __restrict__ d, int n4)
{
  const int i = blockIdx.x * 256 + threadIdx.x;
  if (i >= n4) return;
  const float4 v = ((const float4*)s)[i];
  u16x4 o;
  o[0] = f2bf(v.x); o[1] = f2bf(v.y); o[2] = f2bf(v.z); o[3] = f2bf(v.w);
  ((u16x4*)d)[i] = o;
}

// ---------------------------------------------------------------------------
// RoPE + repack: qkv fp32 (B*S, 4608) ->
//   Q bf16 (B,H,S,D), K bf16 (B,KVH,S,D), V^T bf16 (B,KVH,D,S).
// ---------------------------------------------------------------------------
__global__ __launch_bounds__(256) void rope_pack(
    const float* __restrict__ QKV, const int* __restrict__ pos,
    unsigned short* __restrict__ Q, unsigned short* __restrict__ Kc,
    unsigned short* __restrict__ Vt)
{
  const int bs = blockIdx.x;             // 0 .. B*S-1
  const int b  = bs / S_;
  const int s  = bs - b * S_;
  const float* src = QKV + (size_t)bs * OP_;
  const float p = (float)pos[bs];
  const float kfreq = 0.19188209108283716f;   // ln(10000)/48

  for (int i = threadIdx.x; i < (H_ + KVH_) * 48; i += 256) {
    const int head = i / 48;
    const int f    = i - head * 48;
    const float ang = p * __expf(-kfreq * (float)f);
    float sn, cs;
    __sincosf(ang, &sn, &cs);
    const int srcoff = head * D_;             // q heads then k heads, contiguous
    const float x1 = src[srcoff + f];
    const float x2 = src[srcoff + 48 + f];
    const float o1 = x1 * cs - x2 * sn;
    const float o2 = x2 * cs + x1 * sn;
    if (head < H_) {
      const size_t base = ((size_t)(b * H_ + head) * S_ + s) * D_;
      Q[base + f]      = f2bf(o1);
      Q[base + 48 + f] = f2bf(o2);
    } else {
      const int kvh = head - H_;
      const size_t base = ((size_t)(b * KVH_ + kvh) * S_ + s) * D_;
      Kc[base + f]      = f2bf(o1);
      Kc[base + 48 + f] = f2bf(o2);
    }
  }
  for (int i = threadIdx.x; i < KVH_ * D_; i += 256) {
    const int kvh = i / D_;
    const int d   = i - kvh * D_;
    const float x = src[(H_ + KVH_) * D_ + i];
    Vt[((size_t)(b * KVH_ + kvh) * D_ + d) * S_ + s] = f2bf(x);
  }
}

// ---------------------------------------------------------------------------
// Causal flash attention, 1 wave per 16 query rows of one (b,h).
// 32-key tiles: 6 WMMA Q@K^T (D=96 = 3x K32, two 16-key tiles), online softmax
// (half-wave shfl_xor reductions), P via LDS C-frag->A-frag, 6 WMMA P@V.
// Output written bf16 (feeds O-projection GEMM directly).
// ---------------------------------------------------------------------------
__global__ __launch_bounds__(32) void attn_kernel(
    const unsigned short* __restrict__ Qb,
    const unsigned short* __restrict__ Kb,
    const unsigned short* __restrict__ Vtb,
    unsigned short* __restrict__ Ab)
{
  __shared__ unsigned short Pld[16 * 32];

  const int lane = threadIdx.x;
  const int q0   = blockIdx.x * 16;
  const int h    = blockIdx.y;
  const int b    = blockIdx.z;
  const int kvh  = h >> 2;                       // GQA: 4 Q heads per KV head

  const unsigned short* qp = Qb  + (size_t)(b * H_   + h  ) * S_ * D_;
  const unsigned short* kp = Kb  + (size_t)(b * KVH_ + kvh) * S_ * D_;
  const unsigned short* vp = Vtb + (size_t)(b * KVH_ + kvh) * D_ * S_;

  const int arow = lane & 15;
  const int rowg = lane >> 4;
  const int kb   = rowg * 8;
  const float scale = 0.10206207261596577f;      // 1/sqrt(96)

  v16bf qf[3];
#pragma unroll
  for (int c = 0; c < 3; ++c)
    qf[c] = load_frag(qp + (size_t)(q0 + arow) * D_ + c * 32, kb);

  v8f o[6];
#pragma unroll
  for (int i = 0; i < 6; ++i) o[i] = {};
  float mrow[8], lrow[8];
#pragma unroll
  for (int r = 0; r < 8; ++r) { mrow[r] = -1e30f; lrow[r] = 0.f; }

  for (int key0 = 0; key0 < q0 + 16; key0 += 32) {
    if (key0 + 32 < q0 + 16)   // prefetch next K tile (global_prefetch_b8)
      __builtin_prefetch(kp + (size_t)(key0 + 32 + arow) * D_, 0, 1);

    v8f s0 = {}, s1 = {};
#pragma unroll
    for (int c = 0; c < 3; ++c) {
      v16bf k0f = load_frag(kp + (size_t)(key0      + arow) * D_ + c * 32, kb);
      v16bf k1f = load_frag(kp + (size_t)(key0 + 16 + arow) * D_ + c * 32, kb);
      s0 = wmma_bf16(qf[c], k0f, s0);
      s1 = wmma_bf16(qf[c], k1f, s1);
    }

#pragma unroll
    for (int r = 0; r < 8; ++r) {
      const int m  = r + rowg * 8;
      const int qg = q0 + m;
      const bool ok0 = (key0 + arow)      <= qg;
      const bool ok1 = (key0 + 16 + arow) <= qg;
      const float v0 = s0[r] * scale;
      const float v1 = s1[r] * scale;
      float mx = mrow[r];
      if (ok0) mx = fmaxf(mx, v0);
      if (ok1) mx = fmaxf(mx, v1);
#pragma unroll
      for (int off = 1; off < 16; off <<= 1)
        mx = fmaxf(mx, __shfl_xor(mx, off, 32));
      const float rescale = __expf(mrow[r] - mx);
      const float p0 = ok0 ? __expf(v0 - mx) : 0.f;
      const float p1 = ok1 ? __expf(v1 - mx) : 0.f;
      float rs = p0 + p1;
#pragma unroll
      for (int off = 1; off < 16; off <<= 1)
        rs += __shfl_xor(rs, off, 32);
      mrow[r] = mx;
      lrow[r] = lrow[r] * rescale + rs;
#pragma unroll
      for (int n0 = 0; n0 < 6; ++n0) o[n0][r] *= rescale;
      Pld[m * 32 + arow]      = f2bf(p0);
      Pld[m * 32 + 16 + arow] = f2bf(p1);
    }

    // same-wave DS is in-order; wait for all lanes' P writes before frag read
    asm volatile("s_wait_dscnt 0" ::: "memory");
    v16bf pf = load_frag(&Pld[arow * 32], kb);

#pragma unroll
    for (int n0 = 0; n0 < 6; ++n0) {
      v16bf vf = load_frag(vp + (size_t)(n0 * 16 + arow) * S_ + key0, kb);
      o[n0] = wmma_bf16(pf, vf, o[n0]);
    }
  }

  // normalize + write bf16 attn buffer (B*S, H*D); 32-bit indices
#pragma unroll
  for (int n0 = 0; n0 < 6; ++n0)
#pragma unroll
    for (int r = 0; r < 8; ++r) {
      const int m = r + rowg * 8;
      const int idx = (b * S_ + q0 + m) * HID_ + h * D_ + n0 * 16 + arow;
      Ab[idx] = f2bf(o[n0][r] / lrow[r]);
    }
}

// ---------------------------------------------------------------------------
extern "C" void kernel_launch(void* const* d_in, const int* in_sizes, int n_in,
                              void* d_out, int out_size, void* d_ws, size_t ws_size,
                              hipStream_t stream) {
  (void)in_sizes; (void)n_in; (void)out_size; (void)ws_size;
  const float* hid  = (const float*)d_in[0];
  const int*   pos  = (const int*)d_in[1];
  const float* qkvw = (const float*)d_in[2];
  const float* ow   = (const float*)d_in[3];
  float* out = (float*)d_out;

  char* w = (char*)d_ws;
  float* qkv            = (float*)w;          w += (size_t)B_ * S_ * OP_ * 4;
  unsigned short* hidB  = (unsigned short*)w; w += (size_t)B_ * S_ * HID_ * 2;
  unsigned short* qkvwB = (unsigned short*)w; w += (size_t)OP_ * HID_ * 2;
  unsigned short* owB   = (unsigned short*)w; w += (size_t)HID_ * HID_ * 2;
  unsigned short* attnB = (unsigned short*)w; w += (size_t)B_ * S_ * HID_ * 2;
  unsigned short* Qb    = (unsigned short*)w; w += (size_t)B_ * H_   * S_ * D_ * 2;
  unsigned short* Kb    = (unsigned short*)w; w += (size_t)B_ * KVH_ * S_ * D_ * 2;
  unsigned short* Vt    = (unsigned short*)w; w += (size_t)B_ * KVH_ * S_ * D_ * 2;

  // 0) one-time bf16 conversion of activations + weights
  {
    int n4h = (B_ * S_ * HID_) / 4;
    cvt_f32_bf16<<<(n4h + 255) / 256, 256, 0, stream>>>(hid, hidB, n4h);
    int n4q = (OP_ * HID_) / 4;
    cvt_f32_bf16<<<(n4q + 255) / 256, 256, 0, stream>>>(qkvw, qkvwB, n4q);
    int n4o = (HID_ * HID_) / 4;
    cvt_f32_bf16<<<(n4o + 255) / 256, 256, 0, stream>>>(ow, owB, n4o);
  }

  // 1) QKV projection: (4096x3072) @ (4608x3072)^T -> fp32
  dim3 g1(OP_ / BN, (B_ * S_) / BM);
  gemm_bf16_tdm<<<g1, 256, 2 * STAGE_BYTES, stream>>>(hidB, qkvwB, qkv,
                                                      B_ * S_, OP_, HID_);

  // 2) RoPE + bf16 repack (V transposed)
  rope_pack<<<B_ * S_, 256, 0, stream>>>(qkv, pos, Qb, Kb, Vt);

  // 3) causal flash attention -> bf16
  dim3 ga(S_ / 16, H_, B_);
  attn_kernel<<<ga, 32, 0, stream>>>(Qb, Kb, Vt, attnB);

  // 4) output projection: (4096x3072) @ (3072x3072)^T -> fp32 out
  dim3 g2(HID_ / BN, (B_ * S_) / BM);
  gemm_bf16_tdm<<<g2, 256, 2 * STAGE_BYTES, stream>>>(attnB, owB, out,
                                                      B_ * S_, HID_, HID_);
}
